// MultiHeadedAttentionLayer_63943473103398
// MI455X (gfx1250) — matrix operand order repaired and checked by reading, not compiled, not run on hardware
//
#include <hip/hip_runtime.h>
#include <hip/hip_bf16.h>

// BigBird block-sparse attention for MI455X (gfx1250, wave32, WMMA).
// B=2 H=12 S=4096 D=64 BLK=64 NB=64 R=3.
// Flash-attention over each row-block's key-block list; f16 WMMA, f32 accum.

#define B_   2
#define H_   12
#define S_   4096
#define D_   64
#define BLK_ 64
#define NB_  64
#define R_   3
#define NEG_ (-10000.0f)
#define SCALE_ 0.125f   // 1/sqrt(64)

typedef _Float16 h16v __attribute__((ext_vector_type(16)));
typedef _Float16 h8v  __attribute__((ext_vector_type(8)));
typedef _Float16 h4v  __attribute__((ext_vector_type(4)));
typedef float    f8v  __attribute__((ext_vector_type(8)));

union H16 { h16v v; h8v h[2]; _Float16 e[16]; };
union H4  { h4v  v; _Float16 e[4]; };

// LDS row strides (halfs), padded; 72*2B = 144B keeps 16B alignment (72 % 8 == 0).
#define KROW 72
#define VROW 72
#define PROW 72

__global__ __launch_bounds__(128)
void bigbird_attn_kernel(const float* __restrict__ Q,
                         const float* __restrict__ Kg,
                         const float* __restrict__ Vg,
                         const float* __restrict__ to_mask,
                         const float* __restrict__ from_mask,
                         const int*   __restrict__ rand_attn,
                         float* __restrict__ Out)
{
    __shared__ alignas(16) _Float16 Kbuf[BLK_ * KROW];     // K block, row-major [key][d]
    __shared__ alignas(16) _Float16 Vtb [D_   * VROW];     // V block, transposed [d][key]
    __shared__ alignas(16) _Float16 Pbuf[4 * 16 * PROW];   // per-wave P scratch [q][key]

    const int qb  = blockIdx.x;   // query block 0..63
    const int hh  = blockIdx.y;
    const int bb  = blockIdx.z;
    const int tid = threadIdx.x;
    const int l   = tid & 31;     // lane
    const int w   = tid >> 5;     // wave 0..3
    const int lm  = l & 15;       // lane % 16
    const int hi  = (l >> 4) & 1; // lane half-group

    // ---- key-block list (uniform per workgroup) ----------------------------
    int blist[8];
    int nblk;
    const bool full = (qb == 0) || (qb == NB_ - 1);
    if (full) {
        nblk = NB_;
    } else {
        const int rbase = ((bb * H_ + hh) * (NB_ - 2) + (qb - 1)) * R_;
        const int r0 = rand_attn[rbase + 0];
        const int r1 = rand_attn[rbase + 1];
        const int r2 = rand_attn[rbase + 2];
        if (qb == 1) {
            blist[0] = 0; blist[1] = 1; blist[2] = 2; blist[3] = NB_ - 1;
            blist[4] = r0; blist[5] = r1; blist[6] = r2; nblk = 7;
        } else if (qb == NB_ - 2) {
            blist[0] = 0; blist[1] = NB_ - 3; blist[2] = NB_ - 2; blist[3] = NB_ - 1;
            blist[4] = r0; blist[5] = r1; blist[6] = r2; nblk = 7;
        } else {
            blist[0] = 0; blist[1] = qb - 1; blist[2] = qb; blist[3] = qb + 1;
            blist[4] = NB_ - 1; blist[5] = r0; blist[6] = r1; blist[7] = r2; nblk = 8;
        }
    }

    // ---- Q fragments: 16 rows per wave, A-layout, pre-scaled ---------------
    // A(l,h): M = l%16 ; K = (h/8)*16 + (l/16)*8 + (h%8)  (+ 32*chunk)
    const int  q0   = qb * BLK_ + w * 16;
    const long qrow = ((long)(bb * H_ + hh) * S_ + q0 + lm) * D_;
    H16 qf[2];
#pragma unroll
    for (int c = 0; c < 2; ++c) {
        const float* qp = Q + qrow + c * 32 + hi * 8;
#pragma unroll
        for (int j = 0; j < 8; ++j) qf[c].e[j]     = (_Float16)(qp[j]      * SCALE_);
#pragma unroll
        for (int j = 0; j < 8; ++j) qf[c].e[8 + j] = (_Float16)(qp[16 + j] * SCALE_);
    }

    // ---- online-softmax state ----------------------------------------------
    f8v o[4] = {};          // O tile 16x64 f32, C-layout, 4 d-tiles
    float mrun[8], lrun[8]; // per row m = r + 8*hi (matches C-layout striping)
#pragma unroll
    for (int r = 0; r < 8; ++r) { mrun[r] = -1e30f; lrun[r] = 0.0f; }

    _Float16* pw = &Pbuf[w * 16 * PROW];

    // ---- stream over key blocks --------------------------------------------
    for (int it = 0; it < nblk; ++it) {
        const int  kb    = full ? it : blist[it];
        const long kbase = ((long)(bb * H_ + hh) * S_ + kb * BLK_) * D_;

        // cooperative stage: K -> Kbuf (row-major f16, packed b64 stores),
        //                    V -> Vtb (transposed f16, scatter b16 stores)
#pragma unroll
        for (int j = 0; j < 8; ++j) {
            const int i   = j * 128 + tid;   // 0..1023 float4s
            const int row = i >> 4;          // key in block
            const int c4  = (i & 15) * 4;    // d start
            const float4 k4 = *(const float4*)(Kg + kbase + row * D_ + c4);
            H4 kp4;
            kp4.e[0] = (_Float16)k4.x; kp4.e[1] = (_Float16)k4.y;
            kp4.e[2] = (_Float16)k4.z; kp4.e[3] = (_Float16)k4.w;
            *(h4v*)&Kbuf[row * KROW + c4] = kp4.v;   // one 8B DS store
            const float4 v4 = *(const float4*)(Vg + kbase + row * D_ + c4);
            Vtb[(c4 + 0) * VROW + row] = (_Float16)v4.x;
            Vtb[(c4 + 1) * VROW + row] = (_Float16)v4.y;
            Vtb[(c4 + 2) * VROW + row] = (_Float16)v4.z;
            Vtb[(c4 + 3) * VROW + row] = (_Float16)v4.w;
        }
        __syncthreads();

        // prefetch next key block into cache while we compute on this one
        if (it + 1 < nblk) {
            const int  kbn = full ? (it + 1) : blist[it + 1];
            const long nb  = ((long)(bb * H_ + hh) * S_ + kbn * BLK_) * D_;
            __builtin_prefetch(Kg + nb + tid * 32, 0, 1);  // 128 thr * 128B = 16KB
            __builtin_prefetch(Vg + nb + tid * 32, 0, 1);
        }

        // S = (Q * scale) . K^T : batch 4 B-fragment loads per chunk, then 4 WMMAs
        // B(l,h): N = l%16 (key) ; K(d) = (h/8)*16 + (l/16)*8 + (h%8) (+ 32*chunk)
        f8v s[4] = {};
#pragma unroll
        for (int c = 0; c < 2; ++c) {
            H16 bf[4];
#pragma unroll
            for (int kt = 0; kt < 4; ++kt) {
                const _Float16* kp = &Kbuf[(kt * 16 + lm) * KROW + c * 32 + hi * 8];
                bf[kt].h[0] = *(const h8v*)(kp);
                bf[kt].h[1] = *(const h8v*)(kp + 16);
            }
#pragma unroll
            for (int kt = 0; kt < 4; ++kt) {
                s[kt] = __builtin_amdgcn_wmma_f32_16x16x32_f16(
                            false, qf[c].v, false, bf[kt].v, (short)0, s[kt], false, false);
            }
        }

        // additive key mask (column n = l%16, same value for all 8 rows)
#pragma unroll
        for (int kt = 0; kt < 4; ++kt) {
            const float tm  = to_mask[bb * S_ + kb * BLK_ + kt * 16 + lm];
            const float add = (1.0f - tm) * NEG_;
#pragma unroll
            for (int r = 0; r < 8; ++r) s[kt][r] += add;
        }

        // flash update: rowmax / exp / rowsum / rescale O
#pragma unroll
        for (int r = 0; r < 8; ++r) {
            float v = fmaxf(fmaxf(s[0][r], s[1][r]), fmaxf(s[2][r], s[3][r]));
            v = fmaxf(v, __shfl_xor(v, 1, 16));
            v = fmaxf(v, __shfl_xor(v, 2, 16));
            v = fmaxf(v, __shfl_xor(v, 4, 16));
            v = fmaxf(v, __shfl_xor(v, 8, 16));
            const float mn    = fmaxf(mrun[r], v);
            const float alpha = __expf(mrun[r] - mn);
            mrun[r] = mn;
            float rs = 0.0f;
#pragma unroll
            for (int kt = 0; kt < 4; ++kt) {
                const float p = __expf(s[kt][r] - mn);
                s[kt][r] = p;
                rs += p;
            }
            rs += __shfl_xor(rs, 1, 16);
            rs += __shfl_xor(rs, 2, 16);
            rs += __shfl_xor(rs, 4, 16);
            rs += __shfl_xor(rs, 8, 16);
            lrun[r] = lrun[r] * alpha + rs;
            o[0][r] *= alpha; o[1][r] *= alpha; o[2][r] *= alpha; o[3][r] *= alpha;
        }

        // P (C-layout f32) -> wave-private LDS (row-major f16) -> A-layout frags
#pragma unroll
        for (int kt = 0; kt < 4; ++kt)
#pragma unroll
            for (int r = 0; r < 8; ++r)
                pw[(r + 8 * hi) * PROW + kt * 16 + lm] = (_Float16)s[kt][r];
        __builtin_amdgcn_wave_barrier();
        asm volatile("s_wait_dscnt 0" ::: "memory"); // wave-local DS RAW fence

        H16 pf[2];
#pragma unroll
        for (int c = 0; c < 2; ++c) {
            const _Float16* pp = &pw[lm * PROW + c * 32 + hi * 8];
            pf[c].h[0] = *(const h8v*)(pp);
            pf[c].h[1] = *(const h8v*)(pp + 16);
        }

        // O += P . V : batch 4 V^T fragment loads per chunk, then 4 WMMAs
#pragma unroll
        for (int c = 0; c < 2; ++c) {
            H16 vf[4];
#pragma unroll
            for (int dt = 0; dt < 4; ++dt) {
                const _Float16* vp = &Vtb[(dt * 16 + lm) * VROW + c * 32 + hi * 8];
                vf[dt].h[0] = *(const h8v*)(vp);
                vf[dt].h[1] = *(const h8v*)(vp + 16);
            }
#pragma unroll
            for (int dt = 0; dt < 4; ++dt) {
                o[dt] = __builtin_amdgcn_wmma_f32_16x16x32_f16(
                            false, pf[c].v, false, vf[dt].v, (short)0, o[dt], false, false);
            }
        }
        __syncthreads();
    }

    // ---- finalize: O / l * from_mask, store (B,S,H,D) -----------------------
#pragma unroll
    for (int r = 0; r < 8; ++r) {
        const int   qq  = q0 + r + 8 * hi;
        const float inv = (lrun[r] > 0.0f) ? (1.0f / lrun[r]) : 0.0f;
        const float fm  = from_mask[bb * S_ + qq];
        const float sc  = inv * fm;
        const long  ob  = ((long)(bb * S_ + qq) * H_ + hh) * D_ + lm;
        Out[ob +  0] = o[0][r] * sc;
        Out[ob + 16] = o[1][r] * sc;
        Out[ob + 32] = o[2][r] * sc;
        Out[ob + 48] = o[3][r] * sc;
    }
}

extern "C" void kernel_launch(void* const* d_in, const int* in_sizes, int n_in,
                              void* d_out, int out_size, void* d_ws, size_t ws_size,
                              hipStream_t stream) {
    (void)in_sizes; (void)n_in; (void)out_size; (void)d_ws; (void)ws_size;
    const float* Q         = (const float*)d_in[0];
    const float* K         = (const float*)d_in[1];
    const float* V         = (const float*)d_in[2];
    // d_in[3] band_mask, d_in[6] from_blocked, d_in[7] to_blocked: all-ones in
    // the harness; masking is reproduced via to_mask/from_mask.
    const float* from_mask = (const float*)d_in[4];
    const float* to_mask   = (const float*)d_in[5];
    const int*   rand_attn = (const int*)d_in[8];

    dim3 grid(NB_, H_, B_);  // (query block, head, batch) = 64 x 12 x 2
    bigbird_attn_kernel<<<grid, 128, 0, stream>>>(
        Q, K, V, to_mask, from_mask, rand_attn, (float*)d_out);
}
